// target_encoder_47931835023754
// MI455X (gfx1250) — matrix operand order repaired and checked by reading, not compiled, not run on hardware
//
#include <hip/hip_runtime.h>
#include <math.h>

typedef __attribute__((ext_vector_type(16))) _Float16 v16h;
typedef __attribute__((ext_vector_type(8)))  float    v8f;

#define EPSV 1e-5f

// ---------------------------------------------------------------------------
// WMMA A-fragment K index for 16-bit A 16x32 (ISA 7.12.2):
// lanes 0-15: VGPR0-3 -> K 0..7, VGPR4-7 -> K 16..23 ; lanes 16-31: +8
// ---------------------------------------------------------------------------
__device__ __forceinline__ int a_k_of(int e, int lane) {
    int j = e >> 1, r = e & 1;
    int k = (j < 4) ? (2 * j + r) : (16 + 2 * (j - 4) + r);
    return k + ((lane >> 4) << 3);
}

// B fragment loaders (unused): 16 consecutive floats -> 16 f16 elements
__device__ __forceinline__ v16h load_bf_v4(const float* p) {   // 16B-aligned path
    const float4* q = (const float4*)p;
    float4 a0 = q[0], a1 = q[1], a2 = q[2], a3 = q[3];
    v16h bf;
    bf[0]  = (_Float16)a0.x; bf[1]  = (_Float16)a0.y;
    bf[2]  = (_Float16)a0.z; bf[3]  = (_Float16)a0.w;
    bf[4]  = (_Float16)a1.x; bf[5]  = (_Float16)a1.y;
    bf[6]  = (_Float16)a1.z; bf[7]  = (_Float16)a1.w;
    bf[8]  = (_Float16)a2.x; bf[9]  = (_Float16)a2.y;
    bf[10] = (_Float16)a2.z; bf[11] = (_Float16)a2.w;
    bf[12] = (_Float16)a3.x; bf[13] = (_Float16)a3.y;
    bf[14] = (_Float16)a3.z; bf[15] = (_Float16)a3.w;
    return bf;
}
__device__ __forceinline__ v16h load_bf_b32(const float* p) {  // unaligned (tap shifts)
    v16h bf;
#pragma unroll
    for (int e = 0; e < 16; ++e) bf[e] = (_Float16)p[e];
    return bf;
}

// ---------------------------------------------------------------------------
// weight prep: 1x1 conv weights W[Cout][Cin] -> fragment-ready f16 arena
// ---------------------------------------------------------------------------
__global__ void prep_w1(const float* __restrict__ W, v16h* __restrict__ WF,
                        int Cin, int Kt) {
    int kt = blockIdx.x, mt = blockIdx.y, lane = threadIdx.x;
    int o = mt * 16 + (lane & 15);
    v16h af;
#pragma unroll
    for (int e = 0; e < 16; ++e) {
        int c = kt * 32 + a_k_of(e, lane);
        af[e] = (c < Cin) ? (_Float16)W[o * Cin + c] : (_Float16)0.f;
    }
    WF[((size_t)mt * Kt + kt) * 32 + lane] = af;
}

// temporal conv weights W[O][C][9][1], K = tap*64 + c ordering (C==64)
__global__ void prep_wt(const float* __restrict__ W, v16h* __restrict__ WF, int C) {
    int kt = blockIdx.x, mt = blockIdx.y, lane = threadIdx.x;
    int Kt = gridDim.x;
    int o = mt * 16 + (lane & 15);
    v16h af;
#pragma unroll
    for (int e = 0; e < 16; ++e) {
        int KA = kt * 32 + a_k_of(e, lane);
        int tap = KA >> 6, c = KA & 63;
        af[e] = (_Float16)W[(o * C + c) * 9 + tap];
    }
    WF[((size_t)mt * Kt + kt) * 32 + lane] = af;
}

__global__ void zero_f32(float* __restrict__ p, int n) {
    int i = blockIdx.x * blockDim.x + threadIdx.x;
    if (i < n) p[i] = 0.f;
}

// ---------------------------------------------------------------------------
// data_bn + reshape into padded layout: Y[nm][32 rows][PL]; rows>=3 and
// halo/pad columns zeroed. channel = (m*V + v)*C + c
// ---------------------------------------------------------------------------
__global__ void data_bn_reshape(const float* __restrict__ X, const float* __restrict__ g,
                                const float* __restrict__ b, float* __restrict__ Y,
                                int PL, int HALO, int S, int Vn) {
    int idx = blockIdx.x * blockDim.x + threadIdx.x;
    int total = 32 * 32 * PL;
    if (idx >= total) return;
    int s  = idx % PL;
    int c  = (idx / PL) % 32;
    int nm = idx / (PL * 32);
    float out = 0.f;
    int sv = s - HALO;
    if (c < 3 && sv >= 0 && sv < S) {
        int t = sv / Vn, v = sv % Vn;
        int m = nm & 1, n = nm >> 1;
        float x = X[((((size_t)n * 3 + c) * 300 + t) * Vn + v) * 2 + m];
        int ch = (m * Vn + v) * 3 + c;
        out = x * (g[ch] * rsqrtf(1.f + EPSV)) + b[ch];
    }
    Y[idx] = out;
}

// ---------------------------------------------------------------------------
// 1x1 conv WMMA GEMM, register-blocked 16(M) x 64(N) per wave.
// grid = (SP/64, Cout/16, NM), block 32. All loads unconditional.
// ---------------------------------------------------------------------------
__global__ void conv1x1_wmma(const float* __restrict__ X, const v16h* __restrict__ WF,
                             const float* __restrict__ bias,
                             const float* __restrict__ bng, const float* __restrict__ bnb,
                             float* __restrict__ Y,
                             int Crows, int Cout, int PL, int HALO, int Kt, int accumulate) {
    int lane = threadIdx.x;
    int nm = blockIdx.z;
    int mt = blockIdx.y;
    int n0 = blockIdx.x << 6;
    v8f acc[4] = {};
    const v16h* wf = WF + (size_t)mt * Kt * 32 + lane;
    const float* xb = X + (size_t)nm * Crows * PL + HALO + n0 + (lane & 15);
    size_t rowstep = (size_t)32 * PL;
    for (int kt = 0; kt < Kt; ++kt) {
        v16h af = wf[(size_t)kt * 32];
#pragma unroll
        for (int u = 0; u < 4; ++u) {
            v16h bf;
#pragma unroll
            for (int e = 0; e < 16; ++e) bf[e] = (_Float16)xb[(size_t)a_k_of(e, lane) * PL + u * 16];
            acc[u] = __builtin_amdgcn_wmma_f32_16x16x32_f16(false, af, false, bf,
                                                            (short)0, acc[u], false, false);
        }
        xb += rowstep;
    }
    int mrow = lane & 15, hi = lane >> 4;
    float rs = rsqrtf(1.f + EPSV);
#pragma unroll
    for (int r = 0; r < 8; ++r) {
        int ro = r + (hi << 3);
        int o = mt * 16 + ro;
        float bi = bias[o];
        float sc = bng ? (bng[o] * rs) : 1.f;
        float ad = bng ? bnb[o] : 0.f;
        float* yb = Y + ((size_t)nm * Cout + o) * PL + HALO + n0 + mrow;
#pragma unroll
        for (int u = 0; u < 4; ++u) {
            float v = (acc[u][r] + bi) * sc + ad;
            if (accumulate) yb[u * 16] += v; else yb[u * 16] = v;
        }
    }
}

// ---------------------------------------------------------------------------
// 9x1 temporal conv, implicit WMMA GEMM K=576, 16x64 per wave. Zero halos make
// the temporal boundary implicit -> unconditional b32 loads (shifts break align).
// ---------------------------------------------------------------------------
__global__ void tcn_wmma(const float* __restrict__ X, const v16h* __restrict__ WF,
                         const float* __restrict__ bias,
                         const float* __restrict__ bng, const float* __restrict__ bnb,
                         const float* __restrict__ RES,
                         float* __restrict__ Y,
                         int PLin, int HIN, int Vn,
                         int PLout, int HOUT, int Sout) {
    const int C = 64, Kt = 18;
    int lane = threadIdx.x;
    int nm = blockIdx.z;
    int mt = blockIdx.y;
    int n0 = blockIdx.x << 6;
    v8f acc[4] = {};
    const v16h* wf = WF + (size_t)mt * Kt * 32 + lane;
    for (int kt = 0; kt < Kt; ++kt) {
        const float* xr = X + (size_t)nm * C * PLin + HIN + n0 + (lane & 15);
        v16h af = wf[(size_t)kt * 32];
#pragma unroll
        for (int u = 0; u < 4; ++u) {
            v16h bf;
#pragma unroll
            for (int e = 0; e < 16; ++e) {
                int KB = kt * 32 + a_k_of(e, lane);
                int tap = KB >> 6, cB = KB & 63;
                bf[e] = (_Float16)xr[(size_t)cB * PLin + (tap - 4) * Vn + u * 16];
            }
            acc[u] = __builtin_amdgcn_wmma_f32_16x16x32_f16(false, af, false, bf,
                                                            (short)0, acc[u], false, false);
        }
    }
    int mrow = lane & 15, hi = lane >> 4;
    float rs = rsqrtf(1.f + EPSV);
#pragma unroll
    for (int r = 0; r < 8; ++r) {
        int o = mt * 16 + r + (hi << 3);
        float bi = bias[o];
        float sc = bng[o] * rs;
        float ad = bnb[o];
        const float* rb = RES ? (RES + ((size_t)nm * C + o) * PLin + HIN + n0 + mrow) : nullptr;
        float* yb = Y + ((size_t)nm * C + o) * PLout + HOUT + n0 + mrow;
#pragma unroll
        for (int u = 0; u < 4; ++u) {
            float v = (acc[u][r] + bi) * sc + ad;
            if (rb) v += rb[u * 16];
            v = fmaxf(v, 0.f);
            int s = n0 + u * 16 + mrow;
            if (s < Sout) yb[u * 16] = v;
        }
    }
}

// ---------------------------------------------------------------------------
// attention logits partial: per (channel, nm) block, (Vn,Vn) threads,
// 300-deep reduction, atomicAdd into LOG[nm][Vn][Vn]
// ---------------------------------------------------------------------------
__global__ void attn_partial(const float* __restrict__ Q, const float* __restrict__ K,
                             float* __restrict__ LOG,
                             int T, int Vn, int PL, int HALO) {
    int c = blockIdx.x, nm = blockIdx.y;
    int i = threadIdx.x, j = threadIdx.y;
    const float* Qb = Q + ((size_t)nm * 16 + c) * PL + HALO;
    const float* Kb = K + ((size_t)nm * 16 + c) * PL + HALO;
    float acc = 0.f;
    for (int t = 0; t < T; ++t)
        acc += Qb[t * Vn + i] * Kb[t * Vn + j];
    atomicAdd(&LOG[((size_t)nm * Vn + i) * Vn + j], acc);
}

// softmax over i (rows) per column j, + Afix + PA
__global__ void attn_finish(const float* __restrict__ LOG, const float* __restrict__ Afix,
                            const float* __restrict__ PA, float* __restrict__ ATT,
                            int Vn, float invL) {
    __shared__ float sm[625];
    __shared__ float cmax[25], csum[25];
    int nm = blockIdx.x;
    int i = threadIdx.x, j = threadIdx.y;
    float l = LOG[((size_t)nm * Vn + i) * Vn + j] * invL;
    sm[i * Vn + j] = l;
    __syncthreads();
    if (i == 0) {
        float mx = -1e30f;
        for (int ii = 0; ii < Vn; ++ii) mx = fmaxf(mx, sm[ii * Vn + j]);
        float ss = 0.f;
        for (int ii = 0; ii < Vn; ++ii) ss += __expf(sm[ii * Vn + j] - mx);
        cmax[j] = mx; csum[j] = ss;
    }
    __syncthreads();
    float a = __expf(sm[i * Vn + j] - cmax[j]) / csum[j] + Afix[i * Vn + j] + PA[i * Vn + j];
    ATT[((size_t)nm * Vn + i) * Vn + j] = a;
}

// ---------------------------------------------------------------------------
// Z[nm,c,t,j] = sum_i X[nm,c,t,i] * ATT[nm,i,j]  (padded layout, rows 0..Cin-1)
// ---------------------------------------------------------------------------
__global__ void apply_att(const float* __restrict__ X, const float* __restrict__ ATT,
                          float* __restrict__ Z,
                          int Crows, int Cin, int T, int Vn, int PL, int HALO) {
    __shared__ float sa[625];
    int nm = blockIdx.y;
    for (int t = threadIdx.x; t < Vn * Vn; t += blockDim.x)
        sa[t] = ATT[(size_t)nm * Vn * Vn + t];
    __syncthreads();
    int total = Cin * T * Vn;
    int L = blockIdx.x * blockDim.x + threadIdx.x;
    if (L >= total) return;
    int j = L % Vn;
    int ct = L / Vn;
    int c = ct / T, t = ct - c * T;
    const float* xr = X + ((size_t)nm * Crows + c) * PL + HALO + t * Vn;
    float acc = 0.f;
    for (int i = 0; i < Vn; ++i) acc += xr[i] * sa[i * Vn + j];
    Z[((size_t)nm * Crows + c) * PL + HALO + (size_t)t * Vn + j] = acc;
}

// ---------------------------------------------------------------------------
// gcn epilogue over full padded planes: valid -> relu(BN(Yacc)+DX), else 0
// (this also establishes the zero halos the next TCN depends on)
// ---------------------------------------------------------------------------
__global__ void gcn_epilogue(const float* __restrict__ Yacc, const float* __restrict__ DX,
                             const float* __restrict__ bng, const float* __restrict__ bnb,
                             float* __restrict__ OUT, int PL, int HALO, int S, int total) {
    int idx = blockIdx.x * blockDim.x + threadIdx.x;
    if (idx >= total) return;
    int s = idx % PL;
    int o = (idx / PL) & 63;
    int sv = s - HALO;
    float out = 0.f;
    if (sv >= 0 && sv < S) {
        float v = Yacc[idx] * (bng[o] * rsqrtf(1.f + EPSV)) + bnb[o] + DX[idx];
        out = fmaxf(v, 0.f);
    }
    OUT[idx] = out;
}

// ---------------------------------------------------------------------------
// node remap: Y[nm,c,t,p] = sum_v X[nm,c,t,v]*we[v,p] + be[p]  (PL18 -> PL25)
// ---------------------------------------------------------------------------
__global__ void node_remap(const float* __restrict__ X, const float* __restrict__ we,
                           const float* __restrict__ be, float* __restrict__ Y,
                           int T, int Vin, int Vout,
                           int PLin, int HIN, int PLout, int HOUT) {
    __shared__ float swe[450];
    __shared__ float sbe[25];
    int nm = blockIdx.y;
    for (int t = threadIdx.x; t < Vin * Vout; t += blockDim.x) swe[t] = we[t];
    for (int t = threadIdx.x; t < Vout; t += blockDim.x) sbe[t] = be[t];
    __syncthreads();
    int total = 64 * T * Vout;
    int L = blockIdx.x * blockDim.x + threadIdx.x;
    if (L >= total) return;
    int p = L % Vout;
    int ct = L / Vout;
    int c = ct / T, t = ct - c * T;
    const float* xr = X + ((size_t)nm * 64 + c) * PLin + HIN + t * Vin;
    float acc = sbe[p];
    for (int v = 0; v < Vin; ++v) acc += xr[v] * swe[v * Vout + p];
    Y[((size_t)nm * 64 + c) * PLout + HOUT + (size_t)t * Vout + p] = acc;
}

// ---------------------------------------------------------------------------
// x1[n,ch] = mean over (m, t, p) of dense Z[(n*2+m), ch, 7500]
// ---------------------------------------------------------------------------
__global__ void mean_pool(const float* __restrict__ Z, float* __restrict__ O1, int SP) {
    __shared__ float red[256];
    int bid = blockIdx.x;  // n*64 + ch
    int ch = bid & 63, n = bid >> 6;
    float acc = 0.f;
    for (int m = 0; m < 2; ++m) {
        const float* zb = Z + (((size_t)(n * 2 + m)) * 64 + ch) * SP;
        for (int sp = threadIdx.x; sp < SP; sp += blockDim.x) acc += zb[sp];
    }
    red[threadIdx.x] = acc;
    __syncthreads();
    for (int off = 128; off > 0; off >>= 1) {
        if (threadIdx.x < off) red[threadIdx.x] += red[threadIdx.x + off];
        __syncthreads();
    }
    if (threadIdx.x == 0) O1[bid] = red[0] / (float)(2 * SP);
}

// ===========================================================================
extern "C" void kernel_launch(void* const* d_in, const int* in_sizes, int n_in,
                              void* d_out, int out_size, void* d_ws, size_t ws_size,
                              hipStream_t stream) {
    (void)in_sizes; (void)n_in; (void)out_size; (void)ws_size;
    const int NMB = 32, T = 300, V = 18, VP = 25, CH = 64, INTER = 16;
    // spatial padded to multiple of 64 (N-tile), zeroed halos of 4*Vn
    const int S18 = 5400, SP18 = 5440, H18 = 72, PL18 = 5584;
    const int S25 = 7500, SP25 = 7552, H25 = 100, PL25 = 7752;

    auto in = [&](int i) { return (const float*)d_in[i]; };
    // leaves (setup_inputs dict insertion order):
    // 0:x 1:dbn_g 2:dbn_b | l1.gcn 3..15 | l1.tcn 16..19 | l2.gcn 20..28 |
    // l2.tcn 29..32 | l3.gcn 33..41 | l3.tcn 42..45 | 46:we 47:be |
    // adapt.gcn 48..56 | adapt.tcn 57..60 | 61:A 62:pA
    const float* Afix  = in(61);
    const float* pAfix = in(62);

    float* ws = (float*)d_ws;
    const size_t CAPB = (size_t)NMB * CH * PL25;
    const size_t QCAP = (size_t)NMB * INTER * PL25;
    float* A   = ws;
    float* B   = A + CAPB;
    float* Cb  = B + CAPB;
    float* E   = Cb + CAPB;
    float* F   = E + CAPB;
    float* G   = F + QCAP;
    float* LOG = G + QCAP;
    float* ATT = LOG + 20000;
    v16h* ARENA = (v16h*)(ATT + 20000);
    size_t wfoff = 0;
    float* out_z  = (float*)d_out;
    float* out_x1 = out_z + (size_t)NMB * CH * S25;

    auto alloc_frag = [&](size_t cnt) { v16h* p = ARENA + wfoff; wfoff += cnt; return p; };
    auto prep1 = [&](const float* W, int Cin, int Cout) {
        int Kt = (Cin + 31) / 32, Mt = Cout / 16;
        v16h* p = alloc_frag((size_t)Mt * Kt * 32);
        dim3 g(Kt, Mt);
        prep_w1<<<g, 32, 0, stream>>>(W, p, Cin, Kt);
        return p;
    };
    auto prept = [&](const float* W) {
        v16h* p = alloc_frag((size_t)4 * 18 * 32);
        dim3 g(18, 4);
        prep_wt<<<g, 32, 0, stream>>>(W, p, 64);
        return p;
    };
    auto run_conv = [&](const float* X, v16h* WF, const float* bias,
                        const float* bng, const float* bnb, float* Y,
                        int Crows, int Cout, int PL, int HALO, int SP, int acc) {
        dim3 grid(SP / 64, Cout / 16, NMB);
        conv1x1_wmma<<<grid, 32, 0, stream>>>(X, WF, bias, bng, bnb, Y,
                                              Crows, Cout, PL, HALO, Crows / 32, acc);
    };

    auto run_gcn = [&](const float* X, int Crows, int Cin, int Vn, int S, int SP,
                       int PL, int HALO, int base, const float* Af,
                       const float* dxres, int down, float* Yacc, float* OUT) {
        const float* wa = in(base + 0); const float* ba = in(base + 1);
        const float* wb = in(base + 2); const float* bb = in(base + 3);
        const float* wd = in(base + 4); const float* bd = in(base + 5);
        const float* PA = in(base + 6);
        const float* bng = in(base + 7); const float* bnb = in(base + 8);
        float invL = 1.f / (float)(INTER * T);
        for (int i = 0; i < 3; ++i) {
            v16h* fa = prep1(wa + i * INTER * Cin, Cin, INTER);
            v16h* fb = prep1(wb + i * INTER * Cin, Cin, INTER);
            v16h* fd = prep1(wd + i * CH * Cin, Cin, CH);
            run_conv(X, fa, ba + i * INTER, nullptr, nullptr, F, Crows, INTER, PL, HALO, SP, 0);
            run_conv(X, fb, bb + i * INTER, nullptr, nullptr, G, Crows, INTER, PL, HALO, SP, 0);
            int nlog = NMB * Vn * Vn;
            zero_f32<<<(nlog + 255) / 256, 256, 0, stream>>>(LOG, nlog);
            dim3 pg(INTER, NMB), pb(Vn, Vn);
            attn_partial<<<pg, pb, 0, stream>>>(F, G, LOG, T, Vn, PL, HALO);
            attn_finish<<<NMB, pb, 0, stream>>>(LOG, Af + i * Vn * Vn, PA + i * Vn * Vn,
                                                ATT, Vn, invL);
            int tot = Cin * T * Vn;
            dim3 ag((tot + 255) / 256, NMB);
            apply_att<<<ag, 256, 0, stream>>>(X, ATT, E, Crows, Cin, T, Vn, PL, HALO);
            run_conv(E, fd, bd + i * CH, nullptr, nullptr, Yacc, Crows, CH, PL, HALO, SP,
                     (i > 0) ? 1 : 0);
        }
        const float* DX = dxres;
        if (down) {
            v16h* fdn = prep1(in(base + 9), Cin, CH);
            run_conv(X, fdn, in(base + 10), in(base + 11), in(base + 12), E,
                     Crows, CH, PL, HALO, SP, 0);
            DX = E;
        }
        int total = NMB * CH * PL;
        gcn_epilogue<<<(total + 255) / 256, 256, 0, stream>>>(Yacc, DX, bng, bnb, OUT,
                                                              PL, HALO, S, total);
    };

    auto run_tcn = [&](const float* X, int base, const float* RES, float* Y,
                       int Vn, int PLin, int HIN, int SP,
                       int PLout, int HOUT, int Sout) {
        v16h* ft = prept(in(base + 0));
        dim3 grid(SP / 64, CH / 16, NMB);
        tcn_wmma<<<grid, 32, 0, stream>>>(X, ft, in(base + 1), in(base + 2), in(base + 3),
                                          RES, Y, PLin, HIN, Vn, PLout, HOUT, Sout);
    };

    // ---- data_bn + reshape -> A [32][32 rows][PL18] (zeros elsewhere) ----
    {
        int total = 32 * 32 * PL18;
        data_bn_reshape<<<(total + 255) / 256, 256, 0, stream>>>(
            in(0), in(1), in(2), A, PL18, H18, S18, V);
    }
    // zero E's 32-row region so padded K-rows of the 3-channel z-buffer are 0
    {
        int total = NMB * 32 * PL18;
        zero_f32<<<(total + 255) / 256, 256, 0, stream>>>(E, total);
    }

    // ---- l1: gcn(3->64, V, down) + tcn (no residual) ----
    run_gcn(A, 32, 3, V, S18, SP18, PL18, H18, 3, Afix, nullptr, 1, B, Cb);
    run_tcn(Cb, 16, nullptr, B, V, PL18, H18, SP18, PL18, H18, SP18);

    // ---- l2 (residual) ----
    run_gcn(B, 64, 64, V, S18, SP18, PL18, H18, 20, Afix, B, 0, A, Cb);
    run_tcn(Cb, 29, B, A, V, PL18, H18, SP18, PL18, H18, SP18);

    // ---- l3 (residual) ----
    run_gcn(A, 64, 64, V, S18, SP18, PL18, H18, 33, Afix, A, 0, B, Cb);
    run_tcn(Cb, 42, A, B, V, PL18, H18, SP18, PL18, H18, SP18);

    // ---- adapt: remap V->VP, gcn on VP graph, tcn -> dense d_out ----
    {
        int tot = CH * T * VP;
        dim3 rg((tot + 255) / 256, NMB);
        node_remap<<<rg, 256, 0, stream>>>(B, in(46), in(47), A, T, V, VP,
                                           PL18, H18, PL25, H25);
    }
    run_gcn(A, 64, 64, VP, S25, SP25, PL25, H25, 48, pAfix, A, 0, B, Cb);
    run_tcn(Cb, 57, nullptr, out_z, VP, PL25, H25, SP25, S25, 0, S25);

    // ---- x1 ----
    mean_pool<<<16 * CH, 256, 0, stream>>>(out_z, out_x1, S25);
}
